// LightweightConv1d_35416300322895
// MI455X (gfx1250) — matrix operand order repaired and checked by reading, not compiled, not run on hardware
//
#include <hip/hip_runtime.h>

typedef __attribute__((ext_vector_type(16))) _Float16 v16h;
typedef __attribute__((ext_vector_type(8)))  _Float16 v8h;
typedef __attribute__((ext_vector_type(4)))  _Float16 v4h;
typedef __attribute__((ext_vector_type(8)))  float    v8f;
typedef __attribute__((ext_vector_type(4)))  float    v4f;

#define T_DIM 2048
#define B_DIM 32
#define C_DIM 512
#define K_DIM 31
#define P_PAD 15
#define TILES 16      // 16 x 16 = 256 timesteps per block
#define NWAVE 4       // 4 waves x 16 chans = 64 channels per block
#define NROW  320     // staged rows: 256 + 31 halo + zero pad, 10 passes of 32
#define SROW  328     // f16 row stride/chan: 656 B (164 dw === 36 mod 64) -> conflict-free b128 reads

// Stage x[tchunk-15 .. tchunk+304][b][cblk .. cblk+63] into LDS as f16 [chan][row].
// CHECK=true adds t-bounds predication (only the first/last t-chunk needs it);
// out-of-range rows are zero-filled (keeps WMMA NaN-safe: 0 x garbage -> NaN).
template<bool CHECK>
__device__ __forceinline__ void stage_x(const float* __restrict__ x, _Float16* sx,
                                        int tid, int tchunk, long gcol)
{
    const int q  = tid & 15;        // chan quad: chans 4q..4q+3
    const int rs = tid >> 4;        // row slot 0..7
    const v4f vzero = {0.f, 0.f, 0.f, 0.f};
#pragma unroll
    for (int p = 0; p < NROW / 32; ++p) {
        const int r0 = p * 32 + rs * 4;          // 4 rows r0..r0+3 for this thread
        v4f rv[4];
#pragma unroll
        for (int j = 0; j < 4; ++j) {
            const int t = tchunk - P_PAD + r0 + j;
            if (!CHECK || ((unsigned)t < (unsigned)T_DIM))
                rv[j] = *(const v4f*)(x + (long)t * (B_DIM * C_DIM) + gcol + 4 * q);
            else
                rv[j] = vzero;
        }
        // 4x4 register transpose + f32->f16, one ds_store_b64 per chan
#pragma unroll
        for (int c = 0; c < 4; ++c) {
            v4h hv;
            hv[0] = (_Float16)rv[0][c];
            hv[1] = (_Float16)rv[1][c];
            hv[2] = (_Float16)rv[2][c];
            hv[3] = (_Float16)rv[3][c];
            *(v4h*)(sx + (4 * q + c) * SROW + r0) = hv;
        }
    }
}

// out[t][b][c] = bias[c] + sum_k softmax(w[h])[k] * x[t-15+k][b][c],  h = c/32
__global__ __launch_bounds__(32 * NWAVE)
void lconv_wmma3_kernel(const float* __restrict__ x,
                        const float* __restrict__ w,
                        const float* __restrict__ bias,
                        float* __restrict__ out)
{
    __shared__ _Float16 sx[64 * SROW];     // ~41 KB staged x tile, f16 [chan][row]
    __shared__ float    swt[NWAVE * 32];   // per-wave softmaxed taps

    const int tid   = threadIdx.x;
    const int wave  = tid >> 5;
    const int lane  = tid & 31;
    const int n     = lane & 15;   // tile column (channel); also M row for A-matrix
    const int khalf = lane >> 4;   // which 16-row half of K this lane holds

    const int tchunk = blockIdx.x * (16 * TILES);
    const int b      = blockIdx.y;
    const int cblk   = blockIdx.z * (16 * NWAVE);
    const int c0     = cblk + wave * 16;
    const int h      = c0 >> 5;    // 32 chans/head -> a 16-chan group is head-uniform

    // ---- per-head softmax over the K=31 taps, fp32 (uniform addrs -> scalar path) ----
    float wmax = -3.4e38f;
    for (int k = 0; k < K_DIM; ++k) wmax = fmaxf(wmax, w[h * K_DIM + k]);
    float sum = 0.f;
    for (int k = 0; k < K_DIM; ++k) sum += __expf(w[h * K_DIM + k] - wmax);
    const float inv = 1.f / sum;
    if (lane < K_DIM) swt[wave * 32 + lane] = __expf(w[h * K_DIM + lane] - wmax) * inv;

    // ---- stage the x tile (uniform branch: 6/8 t-chunks take the unpredicated path) ----
    const long gcol = (long)b * C_DIM + cblk;
    const bool interior = (tchunk - P_PAD >= 0) && (tchunk - P_PAD + NROW <= T_DIM);
    if (interior) stage_x<false>(x, sx, tid, tchunk, gcol);
    else          stage_x<true >(x, sx, tid, tchunk, gcol);

    __syncthreads();

    // ---- banded A matrices (16x32 f16), contiguous-K layout (not the ISA table order):
    // lane M = n; V0..7 hold K = 16*khalf+0..15
    const float* sw = &swt[wave * 32];
    auto wgt = [&](int idx) -> _Float16 {
        return (idx >= 0 && idx < K_DIM) ? (_Float16)sw[idx] : (_Float16)0.0f;
    };
    const int m = n;
    v16h A1, A2;
#pragma unroll
    for (int v = 0; v < 8; ++v) {
#pragma unroll
        for (int s = 0; s < 2; ++s) {
            const int Kk = 16 * khalf + 2 * v + s;
            A1[2 * v + s] = wgt(Kk - m);        // fragment rows r in [0,32)
            A2[2 * v + s] = wgt(Kk + 32 - m);   // fragment rows r in [32,64), r<=45 nonzero
        }
    }

    // ---- B fragment: lanes 0-15 hold K=0..15, lanes 16-31 hold K=16..31.
    // [chan][row] f16 LDS puts a lane's 16 K-values in 16 contiguous rows:
    // exactly two aligned ds_load_b128, concatenated register-only.
    const _Float16* myrow = &sx[(unsigned)(wave * 16 + n) * SROW];
    auto loadFrag = [&](int i) -> v16h {
        const int r0 = 16 * i + 16 * khalf;
        const v8h lo = *(const v8h*)(myrow + r0);
        const v8h hi = *(const v8h*)(myrow + r0 + 8);
        return __builtin_shufflevector(lo, hi, 0, 1, 2, 3, 4, 5, 6, 7,
                                       8, 9, 10, 11, 12, 13, 14, 15);
    };

    // rolling 3-fragment buffer: tile j consumes frag(j) [A1] and frag(j+2) [A2]
    v16h frag[3];
    frag[0] = loadFrag(0);
    frag[1] = loadFrag(1);
    frag[2] = frag[0];   // dead value, overwritten at j=0

    const float bv   = bias[c0 + n];
    const long  ocol = gcol + wave * 16 + n;

#pragma unroll
    for (int j = 0; j < TILES; ++j) {
        frag[(j + 2) % 3] = loadFrag(j + 2);

        v8f acc = {};
        acc = __builtin_amdgcn_wmma_f32_16x16x32_f16(
                  false, A1, false, frag[j % 3], (short)0, acc, false, false);
        acc = __builtin_amdgcn_wmma_f32_16x16x32_f16(
                  false, A2, false, frag[(j + 2) % 3], (short)0, acc, false, false);

        // C/D layout: lane n, VGPR i -> row (i + 8*khalf). Output is write-once:
        // non-temporal stores keep x resident in the 192 MB L2 for halo reuse.
        const int trow = tchunk + 16 * j + 8 * khalf;
#pragma unroll
        for (int i = 0; i < 8; ++i) {
            __builtin_nontemporal_store(
                acc[i] + bv, &out[(long)(trow + i) * (B_DIM * C_DIM) + ocol]);
        }
    }
}

extern "C" void kernel_launch(void* const* d_in, const int* in_sizes, int n_in,
                              void* d_out, int out_size, void* d_ws, size_t ws_size,
                              hipStream_t stream)
{
    const float* x    = (const float*)d_in[0];   // (T, B, C) f32
    const float* w    = (const float*)d_in[1];   // (H, 1, K) f32
    const float* bias = (const float*)d_in[2];   // (C,) f32
    float* out        = (float*)d_out;           // (T, B, C) f32

    dim3 grid(T_DIM / (16 * TILES),   // 8 t-chunks of 256
              B_DIM,                  // 32 batches
              C_DIM / (16 * NWAVE));  // 8 channel blocks of 64
    dim3 block(32 * NWAVE);           // 4 wave32 per block
    lconv_wmma3_kernel<<<grid, block, 0, stream>>>(x, w, bias, out);
}
